// XuDecoder_21028159881225
// MI455X (gfx1250) — matrix operand order repaired and checked by reading, not compiled, not run on hardware
//
#include <hip/hip_runtime.h>
#include <hip/hip_bf16.h>

// ---------------------------------------------------------------- constants
#define DS 196     // S
#define DB 64      // B
#define DC 1024    // C
#define DH 1024    // H
#define DE 512     // E
#define DV 32000   // V
#define DT 20      // T

typedef __bf16 bf16_t;
typedef __attribute__((ext_vector_type(16))) __bf16 v16bf;
typedef __attribute__((ext_vector_type(8)))  __bf16 v8bf;
typedef __attribute__((ext_vector_type(8)))  float  v8f;
typedef __attribute__((ext_vector_type(4)))  int    v4i;

// ------------------------------------------------ gfx1250 async-to-LDS path
#if __has_builtin(__builtin_amdgcn_global_load_async_to_lds_b128)
#define HAVE_ASYNC_LDS 1
#warning "CDNA5: using __builtin_amdgcn_global_load_async_to_lds_b128"
typedef v4i __attribute__((address_space(1)))* gv4i_t;   // global 16B chunk
typedef v4i __attribute__((address_space(3)))* lv4i_t;   // LDS 16B chunk
#else
#define HAVE_ASYNC_LDS 0
#warning "CDNA5: async to-LDS builtin NOT available; sync load+ds_store fallback"
#endif

#if __has_builtin(__builtin_amdgcn_s_wait_asynccnt)
#define WAIT_ASYNC() __builtin_amdgcn_s_wait_asynccnt(0)
#else
#define WAIT_ASYNC() asm volatile("s_wait_asynccnt 0x0" ::: "memory")
#endif

// ---------------------------------------------------------------- reductions
__device__ __forceinline__ float block_sum(float v) {
    __shared__ float sm_s[8];
    for (int o = 16; o; o >>= 1) v += __shfl_down(v, o, 32);
    int lane = threadIdx.x & 31, w = threadIdx.x >> 5;
    __syncthreads();
    if (lane == 0) sm_s[w] = v;
    __syncthreads();
    float r = 0.f;
    int nw = (blockDim.x + 31) >> 5;
    for (int i = 0; i < nw; ++i) r += sm_s[i];
    return r;
}

__device__ __forceinline__ float block_max(float v) {
    __shared__ float sm_m[8];
    for (int o = 16; o; o >>= 1) v = fmaxf(v, __shfl_down(v, o, 32));
    int lane = threadIdx.x & 31, w = threadIdx.x >> 5;
    __syncthreads();
    if (lane == 0) sm_m[w] = v;
    __syncthreads();
    float r = -3.4e38f;
    int nw = (blockDim.x + 31) >> 5;
    for (int i = 0; i < nw; ++i) r = fmaxf(r, sm_m[i]);
    return r;
}

// ---------------------------------------------------------------- WMMA GEMM
// C[M,N] = A[M,K](bf16, row-major, lda) * Bt[N,K]^T (bf16, row-major, ldb) + bias
// Block: 8 waves -> 16(M) x 128(N) tile; A tile staged once into LDS and
// shared by all waves (async to-LDS when available); B streamed from global.
// grid = (N/128, M/16). K mult of 32, N mult of 128, M mult of 16.
// dynamic LDS = 16*K*2 bytes (<= 48KB for K=1536).
__global__ void __launch_bounds__(256)
k_wmma_gemm(const bf16_t* __restrict__ A, int lda,
            const bf16_t* __restrict__ Bt, int ldb,
            const float* __restrict__ bias,
            float* __restrict__ Cf, bf16_t* __restrict__ Cb, int ldc,
            int K)
{
    extern __shared__ char smem_raw[];
    bf16_t* Abuf = (bf16_t*)smem_raw;                 // [16][K]

    const int lane = threadIdx.x & 31;
    const int wave = threadIdx.x >> 5;
    const int m0   = blockIdx.y << 4;
    const int n0   = (blockIdx.x * 8 + wave) << 4;
    const int half = lane >> 4;      // 0: lanes 0-15, 1: lanes 16-31
    const int l15  = lane & 15;

    // ---- stage A tile [16, K] into LDS (16B chunks, no divergence:
    //      total = 16*K/8 is a multiple of 256 for K in {512,1024,1536})
    {
        const int total = (16 * K) >> 3;              // # of 8-element chunks
        for (int c = threadIdx.x; c < total; c += 256) {
            const int el  = c << 3;
            const int row = el / K;
            const int col = el - row * K;
            const bf16_t* gsrc = A + (size_t)(m0 + row) * lda + col;
            bf16_t*       ldst = Abuf + row * K + col;
#if HAVE_ASYNC_LDS
            __builtin_amdgcn_global_load_async_to_lds_b128(
                (gv4i_t)(void*)gsrc, (lv4i_t)(void*)ldst, 0, 0);
#else
            *(v8bf*)ldst = *(const v8bf*)gsrc;
#endif
        }
#if HAVE_ASYNC_LDS
        WAIT_ASYNC();
#endif
        __syncthreads();
    }

    // A fragment (16-bit A 16x32 layout): lane(m=l15, half) holds
    //   elems 0..7  -> K = half*8 + 0..7 ; elems 8..15 -> K = 16+half*8+0..7
    const bf16_t* alrow = Abuf + l15 * K + half * 8;  // LDS
    // B fragment (32x16): lane(n=l15, half) holds K = half*8 + 0..7, 16+half*8 + 0..7 of row n
    const bf16_t* brow = Bt + (size_t)(n0 + l15) * ldb + half * 8;

    v8f acc = {};
    for (int k0 = 0; k0 < K; k0 += 32) {
        __builtin_prefetch((const void*)(brow + k0 + 128), 0, 1);
        v8bf a_lo = *(const v8bf*)(alrow + k0);       // ds_load_b128
        v8bf a_hi = *(const v8bf*)(alrow + k0 + 16);
        v8bf b_lo = *(const v8bf*)(brow + k0);
        v8bf b_hi = *(const v8bf*)(brow + k0 + 16);
        v16bf av = __builtin_shufflevector(a_lo, a_hi,
                     0,1,2,3,4,5,6,7,8,9,10,11,12,13,14,15);
        v16bf bv = __builtin_shufflevector(b_lo, b_hi,
                     0,1,2,3,4,5,6,7,8,9,10,11,12,13,14,15);
        acc = __builtin_amdgcn_wmma_f32_16x16x32_bf16(
                  false, av, false, bv, (short)0, acc, false, false);
    }

    // C/D layout: acc[j] -> M = half*8 + j, N = l15
    const int n = n0 + l15;
    const float bia = bias ? bias[n] : 0.f;
    #pragma unroll
    for (int j = 0; j < 8; ++j) {
        const int m = m0 + half * 8 + j;
        const float v = acc[j] + bia;
        if (Cf) Cf[(size_t)m * ldc + n] = v;
        else    Cb[(size_t)m * ldc + n] = (bf16_t)v;
    }
}

// ---------------------------------------------------------------- elementwise
__global__ void k_cvt_bf16(const float* __restrict__ in, bf16_t* __restrict__ out, int n) {
    for (int i = blockIdx.x * blockDim.x + threadIdx.x; i < n; i += gridDim.x * blockDim.x)
        out[i] = (bf16_t)in[i];
}

// out[c, r] = in[r, c]  (fp32 [rows, cols] -> bf16 [cols, rows])
__global__ void k_cvt_bf16_T(const float* __restrict__ in, bf16_t* __restrict__ out,
                             int rows, int cols) {
    int n = rows * cols;
    for (int i = blockIdx.x * blockDim.x + threadIdx.x; i < n; i += gridDim.x * blockDim.x) {
        int r = i / cols, c = i - r * cols;
        out[(size_t)c * rows + r] = (bf16_t)in[i];
    }
}

__global__ void k_embed(const int* __restrict__ y, const float* __restrict__ embW,
                        float* __restrict__ yemb, int n) {   // n = T*B*E
    for (int i = blockIdx.x * blockDim.x + threadIdx.x; i < n; i += gridDim.x * blockDim.x) {
        int e  = i % DE;
        int tb = i / DE;                  // t*B + b
        int tok = y[tb];
        yemb[i] = embW[(size_t)tok * DE + e];
    }
}

__global__ void k_init(float* __restrict__ h, bf16_t* __restrict__ hbf,
                       float* __restrict__ out) {
    int i = blockIdx.x * blockDim.x + threadIdx.x;
    if (i < DB * DH) { h[i] = 0.f; hbf[i] = (bf16_t)0.f; }
    if (i == 0) out[0] = 0.f;
}

// scores[s*B+b] = sum_c tanh(ctxp_bf[sb, c] + hidp[b, c]) * mlp[c]
// one block (128 threads) per sb; vector loads of 8 bf16
__global__ void __launch_bounds__(128)
k_att_scores(const bf16_t* __restrict__ ctxp, const float* __restrict__ hidp,
             const float* __restrict__ mlp, float* __restrict__ scores) {
    const int sb = blockIdx.x;
    const int b  = sb & (DB - 1);
    const bf16_t* row = ctxp + (size_t)sb * DC;
    const float*  hp  = hidp + (size_t)b * DC;
    float p = 0.f;
    for (int c8 = threadIdx.x << 3; c8 < DC; c8 += blockDim.x << 3) {
        v8bf rv = *(const v8bf*)(row + c8);
        #pragma unroll
        for (int j = 0; j < 8; ++j)
            p += tanhf((float)rv[j] + hp[c8 + j]) * mlp[c8 + j];
    }
    p = block_sum(p);
    if (threadIdx.x == 0) scores[sb] = p;
}

// softmax over S per batch column; one block per b
__global__ void __launch_bounds__(256)
k_softmax_s(const float* __restrict__ scores, float* __restrict__ alpha) {
    const int b = blockIdx.x;
    const int s = threadIdx.x;
    float v = (s < DS) ? scores[s * DB + b] : -3.4e38f;
    float m = block_max(v);
    float e = (s < DS) ? __expf(v - m) : 0.f;
    float sum = block_sum(e);
    if (s < DS) alpha[s * DB + b] = e / sum;
}

// z[b, c8..c8+7] = sum_s alpha[s*B+b] * ctx_bf[(s*B+b)*C + c8..]; bf16 reads
__global__ void k_weighted_ctx(const float* __restrict__ alpha,
                               const bf16_t* __restrict__ ctxb,
                               float* __restrict__ z) {
    int i = blockIdx.x * blockDim.x + threadIdx.x;    // i over B*C/8
    if (i >= DB * DC / 8) return;
    int b  = i / (DC / 8);
    int c8 = (i - b * (DC / 8)) << 3;
    float acc[8];
    #pragma unroll
    for (int j = 0; j < 8; ++j) acc[j] = 0.f;
    for (int s = 0; s < DS; ++s) {
        float a = alpha[s * DB + b];
        v8bf cv = *(const v8bf*)(ctxb + ((size_t)s * DB + b) * DC + c8);
        #pragma unroll
        for (int j = 0; j < 8; ++j) acc[j] += a * (float)cv[j];
    }
    #pragma unroll
    for (int j = 0; j < 8; ++j) z[b * DC + c8 + j] = acc[j];
}

// x_bf[b, 0:E] = yemb_t ; x_bf[b, E:E+C] = z
__global__ void k_build_x(const float* __restrict__ yemb_t, const float* __restrict__ z,
                          bf16_t* __restrict__ xbf) {
    int i = blockIdx.x * blockDim.x + threadIdx.x;    // i over B*(E+C)
    if (i >= DB * (DE + DC)) return;
    int b = i / (DE + DC), j = i - b * (DE + DC);
    float v = (j < DE) ? yemb_t[b * DE + j] : z[b * DC + (j - DE)];
    xbf[i] = (bf16_t)v;
}

// GRU gates; gi/gh are [B, 3H] with bias already added
__global__ void k_gru(const float* __restrict__ gi, const float* __restrict__ gh,
                      float* __restrict__ h, bf16_t* __restrict__ hbf) {
    int i = blockIdx.x * blockDim.x + threadIdx.x;    // i over B*H
    if (i >= DB * DH) return;
    int b = i >> 10, j = i & (DH - 1);
    const float* gib = gi + (size_t)b * 3 * DH;
    const float* ghb = gh + (size_t)b * 3 * DH;
    float i_r = gib[j], i_z = gib[DH + j], i_n = gib[2 * DH + j];
    float h_r = ghb[j], h_z = ghb[DH + j], h_n = ghb[2 * DH + j];
    float r = 1.f / (1.f + __expf(-(i_r + h_r)));
    float zz = 1.f / (1.f + __expf(-(i_z + h_z)));
    float nn = tanhf(i_n + r * h_n);
    float hv = (1.f - zz) * nn + zz * h[i];
    h[i] = hv;
    hbf[i] = (bf16_t)hv;
}

// logit fuse: lg = tanh( tanh(tmp1) + yemb_t + tmp2 ) -> bf16
__global__ void k_logit(const float* __restrict__ tmp1, const float* __restrict__ tmp2,
                        const float* __restrict__ yemb_t, bf16_t* __restrict__ lgbf) {
    int i = blockIdx.x * blockDim.x + threadIdx.x;    // i over B*E
    if (i >= DB * DE) return;
    float l = tanhf(tmp1[i]) + yemb_t[i] + tmp2[i];
    lgbf[i] = (bf16_t)tanhf(l);
}

// per-row masked NLL over logits [B, V]; one block per b; atomicAdd into out
__global__ void __launch_bounds__(256)
k_nll(const float* __restrict__ logits, const int* __restrict__ y_next,
      float* __restrict__ out) {
    const int b = blockIdx.x;
    const float* row = logits + (size_t)b * DV;
    float m = -3.4e38f;
    for (int v = threadIdx.x; v < DV; v += blockDim.x) m = fmaxf(m, row[v]);
    m = block_max(m);
    float s = 0.f;
    for (int v = threadIdx.x; v < DV; v += blockDim.x) s += __expf(row[v] - m);
    s = block_sum(s);
    if (threadIdx.x == 0) {
        int tgt = y_next[b];
        if (tgt != 0) {
            float logp = row[tgt] - m - __logf(s);
            atomicAdd(out, -logp);
        }
    }
}

// ---------------------------------------------------------------- host
extern "C" void kernel_launch(void* const* d_in, const int* in_sizes, int n_in,
                              void* d_out, int out_size, void* d_ws, size_t ws_size,
                              hipStream_t stream) {
    (void)in_sizes; (void)n_in; (void)out_size; (void)ws_size;

    const float* ctx        = (const float*)d_in[0];
    const int*   y          = (const int*)d_in[1];
    const float* emb_W      = (const float*)d_in[2];
    const float* att_c2c    = (const float*)d_in[3];
    const float* att_h2c    = (const float*)d_in[4];
    const float* att_mlp    = (const float*)d_in[5];
    const float* gru_W_ih   = (const float*)d_in[6];
    const float* gru_b_ih   = (const float*)d_in[7];
    const float* gru_W_hh   = (const float*)d_in[8];
    const float* gru_b_hh   = (const float*)d_in[9];
    const float* hid2out_W  = (const float*)d_in[10];
    const float* hid2out_b  = (const float*)d_in[11];
    const float* out_ctx_W  = (const float*)d_in[12];
    const float* out_ctx_b  = (const float*)d_in[13];
    const float* out2prob_W = (const float*)d_in[14];
    const float* out2prob_b = (const float*)d_in[15];
    float* out = (float*)d_out;

    // ---- workspace carving (256B aligned)
    size_t off = 0;
    char* base = (char*)d_ws;
    auto carve = [&](size_t bytes) -> void* {
        void* p = base + off;
        off = (off + bytes + 255) & ~(size_t)255;
        return p;
    };
    bf16_t* ctx_bf   = (bf16_t*)carve((size_t)DS * DB * DC * 2);
    bf16_t* ctxp_bf  = (bf16_t*)carve((size_t)DS * DB * DC * 2);
    bf16_t* c2cT_bf  = (bf16_t*)carve((size_t)DC * DC * 2);
    bf16_t* h2cT_bf  = (bf16_t*)carve((size_t)DC * DH * 2);
    bf16_t* wih_bf   = (bf16_t*)carve((size_t)3 * DH * (DE + DC) * 2);
    bf16_t* whh_bf   = (bf16_t*)carve((size_t)3 * DH * DH * 2);
    bf16_t* h2o_bf   = (bf16_t*)carve((size_t)DE * DH * 2);
    bf16_t* octx_bf  = (bf16_t*)carve((size_t)DE * DC * 2);
    bf16_t* o2p_bf   = (bf16_t*)carve((size_t)DV * DE * 2);
    float*  yemb     = (float*) carve((size_t)DT * DB * DE * 4);
    float*  h        = (float*) carve((size_t)DB * DH * 4);
    bf16_t* h_bf     = (bf16_t*)carve((size_t)DB * DH * 2);
    float*  hidp     = (float*) carve((size_t)DB * DC * 4);
    float*  scores   = (float*) carve((size_t)DS * DB * 4);
    float*  alpha    = (float*) carve((size_t)DS * DB * 4);
    float*  z        = (float*) carve((size_t)DB * DC * 4);
    bf16_t* x_bf     = (bf16_t*)carve((size_t)DB * (DE + DC) * 2);
    float*  gi       = (float*) carve((size_t)DB * 3 * DH * 4);
    float*  gh       = (float*) carve((size_t)DB * 3 * DH * 4);
    float*  tmp1     = (float*) carve((size_t)DB * DE * 4);
    float*  tmp2     = (float*) carve((size_t)DB * DE * 4);
    bf16_t* lg_bf    = (bf16_t*)carve((size_t)DB * DE * 2);
    float*  logits   = (float*) carve((size_t)DB * DV * 4);

    auto gemm = [&](const bf16_t* A, int lda, const bf16_t* Bt, int ldb,
                    const float* bias, float* Cf, bf16_t* Cb, int ldc,
                    int M, int N, int K) {
        dim3 g(N / 128, M / 16);
        size_t sh = (size_t)16 * K * 2;               // A tile in LDS
        k_wmma_gemm<<<g, 256, sh, stream>>>(A, lda, Bt, ldb, bias, Cf, Cb, ldc, K);
    };

    // ---- one-time prep
    k_init<<<(DB * DH + 255) / 256, 256, 0, stream>>>(h, h_bf, out);
    k_cvt_bf16_T<<<2048, 256, 0, stream>>>(att_c2c, c2cT_bf, DC, DC);
    k_cvt_bf16_T<<<2048, 256, 0, stream>>>(att_h2c, h2cT_bf, DH, DC);
    k_cvt_bf16<<<4096, 256, 0, stream>>>(gru_W_ih, wih_bf, 3 * DH * (DE + DC));
    k_cvt_bf16<<<4096, 256, 0, stream>>>(gru_W_hh, whh_bf, 3 * DH * DH);
    k_cvt_bf16<<<2048, 256, 0, stream>>>(hid2out_W, h2o_bf, DE * DH);
    k_cvt_bf16<<<2048, 256, 0, stream>>>(out_ctx_W, octx_bf, DE * DC);
    k_cvt_bf16<<<8192, 256, 0, stream>>>(out2prob_W, o2p_bf, DV * DE);
    k_cvt_bf16<<<8192, 256, 0, stream>>>(ctx, ctx_bf, DS * DB * DC);
    k_embed<<<4096, 256, 0, stream>>>(y, emb_W, yemb, DT * DB * DE);

    // ctx_proj = ctx @ att_ctx2ctx  -> bf16 [S*B, C]
    gemm(ctx_bf, DC, c2cT_bf, DC, nullptr, nullptr, ctxp_bf, DC,
         DS * DB, DC, DC);

    // ---- decode loop (19 steps)
    for (int t = 0; t < DT - 1; ++t) {
        const float* yemb_t = yemb + (size_t)t * DB * DE;
        const int*   y_next = y + (size_t)(t + 1) * DB;

        // hidp = h @ att_hid2ctx  [B, C]
        gemm(h_bf, DH, h2cT_bf, DH, nullptr, hidp, nullptr, DC, DB, DC, DH);
        // attention
        k_att_scores<<<DS * DB, 128, 0, stream>>>(ctxp_bf, hidp, att_mlp, scores);
        k_softmax_s<<<DB, 256, 0, stream>>>(scores, alpha);
        k_weighted_ctx<<<(DB * DC / 8 + 255) / 256, 256, 0, stream>>>(alpha, ctx_bf, z);
        // x = [y_emb | z] bf16
        k_build_x<<<(DB * (DE + DC) + 255) / 256, 256, 0, stream>>>(yemb_t, z, x_bf);
        // GRU GEMMs (+bias)
        gemm(x_bf, DE + DC, wih_bf, DE + DC, gru_b_ih, gi, nullptr, 3 * DH,
             DB, 3 * DH, DE + DC);
        gemm(h_bf, DH, whh_bf, DH, gru_b_hh, gh, nullptr, 3 * DH,
             DB, 3 * DH, DH);
        k_gru<<<(DB * DH + 255) / 256, 256, 0, stream>>>(gi, gh, h, h_bf);
        // output heads
        gemm(h_bf, DH, h2o_bf, DH, hid2out_b, tmp1, nullptr, DE, DB, DE, DH);
        gemm(x_bf + DE, DE + DC, octx_bf, DC, out_ctx_b, tmp2, nullptr, DE,
             DB, DE, DC);   // A = z slice of x_bf (bf16), lda = E+C
        k_logit<<<(DB * DE + 255) / 256, 256, 0, stream>>>(tmp1, tmp2, yemb_t, lg_bf);
        // vocab projection [B, V]
        gemm(lg_bf, DE, o2p_bf, DE, out2prob_b, logits, nullptr, DV, DB, DV, DE);
        // masked NLL, accumulate
        k_nll<<<DB, 256, 0, stream>>>(logits, y_next, out);
    }
}
